// TransformerEncoder_72679436583416
// MI455X (gfx1250) — matrix operand order repaired and checked by reading, not compiled, not run on hardware
//
#include <hip/hip_runtime.h>
#include <math.h>

// ---------------------------------------------------------------- types
typedef __attribute__((ext_vector_type(16))) _Float16 v16h;
typedef __attribute__((ext_vector_type(8)))  _Float16 v8h;
typedef __attribute__((ext_vector_type(8)))  float    v8f;

#define BM 128
#define BN 64
#define BK 32
#define LDSW 40   // f16 stride: 80B row pitch, 16B aligned, conflict-free

// ================================================================ GEMM
// C[M,N] = alpha * A[M,K] @ B[K,N] + bias (optional relu)
// TRANSB: B stored as [N,K] row-major (used for Q @ K^T).
// Batched over blockIdx.z with (b,h) decomposition: z = zb*Hd + zh.
template<bool TRANSB, bool RELU>
__global__ __launch_bounds__(256, 2)
void gemm_wmma(const float* __restrict__ A, const float* __restrict__ B,
               const float* __restrict__ bias, float* __restrict__ C,
               int K, int lda, int ldb, int ldc, int Hd,
               long sAb, long sAh, long sBb, long sBh, long sCb, long sCh,
               float alpha)
{
    __shared__ _Float16 sA[BM * LDSW];
    __shared__ _Float16 sB[BN * LDSW];

    const int t    = threadIdx.x;
    const int lane = t & 31;           // wave32
    const int wave = t >> 5;           // 8 waves
    const int wm   = wave & 3;         // 4 waves along M
    const int wn   = wave >> 2;        // 2 waves along N
    const int lr   = lane & 15;
    const int lh   = lane >> 4;

    const int mBase = blockIdx.x * BM;
    const int nBase = blockIdx.y * BN;
    const int z  = blockIdx.z;
    const int zb = z / Hd, zh = z % Hd;
    A += (size_t)zb * sAb + (size_t)zh * sAh;
    B += (size_t)zb * sBb + (size_t)zh * sBh;
    C += (size_t)zb * sCb + (size_t)zh * sCh;

    // cooperative loader mappings
    const int aRow = t >> 1;                       // 0..127
    const int aK   = (t & 1) << 4;                 // 0 / 16
    const int bN   = TRANSB ? (t >> 2) : ((t & 15) << 2);
    const int bK   = TRANSB ? ((t & 3) << 3) : (t >> 4);

    float aReg[16];
    float bReg[8];

    auto globalLoad = [&](int k0) {
        const float* ga = A + (size_t)(mBase + aRow) * lda + (k0 + aK);
        #pragma unroll
        for (int i = 0; i < 4; ++i) {
            const float4 f = *reinterpret_cast<const float4*>(ga + 4 * i);
            aReg[4*i+0] = f.x; aReg[4*i+1] = f.y;
            aReg[4*i+2] = f.z; aReg[4*i+3] = f.w;
        }
        if (TRANSB) {
            const float* gb = B + (size_t)(nBase + bN) * ldb + (k0 + bK);
            #pragma unroll
            for (int i = 0; i < 2; ++i) {
                const float4 f = *reinterpret_cast<const float4*>(gb + 4 * i);
                bReg[4*i+0] = f.x; bReg[4*i+1] = f.y;
                bReg[4*i+2] = f.z; bReg[4*i+3] = f.w;
            }
        } else {
            const float* gb = B + (size_t)(k0 + bK) * ldb + (nBase + bN);
            const float4 f0 = *reinterpret_cast<const float4*>(gb);
            const float4 f1 = *reinterpret_cast<const float4*>(gb + (size_t)16 * ldb);
            bReg[0] = f0.x; bReg[1] = f0.y; bReg[2] = f0.z; bReg[3] = f0.w;
            bReg[4] = f1.x; bReg[5] = f1.y; bReg[6] = f1.z; bReg[7] = f1.w;
        }
    };

    auto ldsStore = [&]() {
        _Float16* da = &sA[aRow * LDSW + aK];
        #pragma unroll
        for (int i = 0; i < 16; ++i) da[i] = (_Float16)aReg[i];
        if (TRANSB) {
            _Float16* db = &sB[bN * LDSW + bK];
            #pragma unroll
            for (int i = 0; i < 8; ++i) db[i] = (_Float16)bReg[i];
        } else {
            #pragma unroll
            for (int j = 0; j < 4; ++j) {
                sB[(bN + j) * LDSW + bK]      = (_Float16)bReg[j];
                sB[(bN + j) * LDSW + bK + 16] = (_Float16)bReg[4 + j];
            }
        }
    };

    v8f acc[2][2] = {};

    const int KT = K / BK;
    globalLoad(0);
    ldsStore();
    __syncthreads();

    for (int kt = 0; kt < KT; ++kt) {
        if (kt + 1 < KT) globalLoad((kt + 1) * BK);   // prefetch next K-tile

        // Fragments per ISA layout: A lane L holds row L&15; v0-3 = K[8h..8h+7],
        // v4-7 = K[16+8h..16+8h+7]. B lane holds col L&15, same K order as A.
        v16h af[2], bf[2];
        #pragma unroll
        for (int mi = 0; mi < 2; ++mi) {
            const int row = wm * 32 + mi * 16 + lr;
            const v8h lo = *reinterpret_cast<const v8h*>(&sA[row * LDSW + 8 * lh]);
            const v8h hi = *reinterpret_cast<const v8h*>(&sA[row * LDSW + 16 + 8 * lh]);
            af[mi] = __builtin_shufflevector(lo, hi, 0,1,2,3,4,5,6,7,8,9,10,11,12,13,14,15);
        }
        #pragma unroll
        for (int ni = 0; ni < 2; ++ni) {
            const int col = wn * 32 + ni * 16 + lr;
            const v8h p0 = *reinterpret_cast<const v8h*>(&sB[col * LDSW + 8 * lh]);
            const v8h p1 = *reinterpret_cast<const v8h*>(&sB[col * LDSW + 16 + 8 * lh]);
            bf[ni] = __builtin_shufflevector(p0, p1, 0,1,2,3,4,5,6,7,8,9,10,11,12,13,14,15);
        }
        #pragma unroll
        for (int mi = 0; mi < 2; ++mi)
            #pragma unroll
            for (int ni = 0; ni < 2; ++ni)
                acc[mi][ni] = __builtin_amdgcn_wmma_f32_16x16x32_f16(
                    false, af[mi], false, bf[ni], (short)0, acc[mi][ni], false, false);

        __syncthreads();
        if (kt + 1 < KT) { ldsStore(); __syncthreads(); }
    }

    // epilogue: C/D layout — lane col = lr (+half via lh), VGPR j = row 8*lh+j
    #pragma unroll
    for (int mi = 0; mi < 2; ++mi) {
        #pragma unroll
        for (int ni = 0; ni < 2; ++ni) {
            const int col = nBase + wn * 32 + ni * 16 + lr;
            const float bv = bias ? bias[col] : 0.0f;
            #pragma unroll
            for (int j = 0; j < 8; ++j) {
                const int row = mBase + wm * 32 + mi * 16 + lh * 8 + j;
                float val = alpha * acc[mi][ni][j] + bv;
                if (RELU) val = fmaxf(val, 0.0f);
                C[(size_t)row * ldc + col] = val;
            }
        }
    }
}

template<bool TB, bool RELU>
static inline void launch_gemm(hipStream_t st, const float* A, const float* B,
                               const float* bias, float* C, int M, int N, int K,
                               int lda, int ldb, int ldc, int Z, int Hd,
                               long sAb, long sAh, long sBb, long sBh,
                               long sCb, long sCh, float alpha)
{
    dim3 grid(M / BM, N / BN, Z);
    gemm_wmma<TB, RELU><<<grid, 256, 0, st>>>(A, B, bias, C, K, lda, ldb, ldc, Hd,
                                              sAb, sAh, sBb, sBh, sCb, sCh, alpha);
}

// ===================================================== embedding + PE
__global__ __launch_bounds__(256)
void embed_pe(const int* __restrict__ ids, const float* __restrict__ emb,
              float* __restrict__ h)
{
    const int row = blockIdx.x;           // b*1024 + s
    const int s   = row & 1023;
    const int id  = ids[row];
    #pragma unroll
    for (int i = 0; i < 4; ++i) {
        const int d  = threadIdx.x + 256 * i;
        const int i2 = d & ~1;
        const float div = expf((float)i2 * (-9.2103403719761836f / 1024.0f)); // ln(1e4)
        const float ang = (float)s * div;
        const float pe  = (d & 1) ? cosf(ang) : sinf(ang);
        h[(size_t)row * 1024 + d] = emb[(size_t)id * 1024 + d] * 32.0f + pe;
    }
}

// ===================================================== in-place softmax
__global__ __launch_bounds__(256)
void softmax_rows(float* __restrict__ p)
{
    __shared__ float red[256];
    float* r = p + (size_t)blockIdx.x * 1024;
    const int t = threadIdx.x;
    float v[4];
    float m = -3.4e38f;
    #pragma unroll
    for (int i = 0; i < 4; ++i) { v[i] = r[t + 256 * i]; m = fmaxf(m, v[i]); }
    red[t] = m; __syncthreads();
    for (int s = 128; s > 0; s >>= 1) { if (t < s) red[t] = fmaxf(red[t], red[t + s]); __syncthreads(); }
    m = red[0]; __syncthreads();
    float sum = 0.0f;
    #pragma unroll
    for (int i = 0; i < 4; ++i) { v[i] = __expf(v[i] - m); sum += v[i]; }
    red[t] = sum; __syncthreads();
    for (int s = 128; s > 0; s >>= 1) { if (t < s) red[t] += red[t + s]; __syncthreads(); }
    const float inv = 1.0f / red[0];
    #pragma unroll
    for (int i = 0; i < 4; ++i) r[t + 256 * i] = v[i] * inv;
}

// ===================================================== residual + LayerNorm
__global__ __launch_bounds__(256)
void add_layernorm(const float* __restrict__ x, const float* __restrict__ rres,
                   const float* __restrict__ g, const float* __restrict__ b,
                   float* __restrict__ out)
{
    __shared__ float red[256];
    const size_t row = blockIdx.x;
    const int t = threadIdx.x;
    float v[4];
    float s = 0.0f;
    #pragma unroll
    for (int i = 0; i < 4; ++i) {
        const int d = t + 256 * i;
        v[i] = x[row * 1024 + d] + rres[row * 1024 + d];
        s += v[i];
    }
    red[t] = s; __syncthreads();
    for (int st = 128; st > 0; st >>= 1) { if (t < st) red[t] += red[t + st]; __syncthreads(); }
    const float mean = red[0] * (1.0f / 1024.0f); __syncthreads();
    float s2 = 0.0f;
    #pragma unroll
    for (int i = 0; i < 4; ++i) { const float dv = v[i] - mean; s2 += dv * dv; }
    red[t] = s2; __syncthreads();
    for (int st = 128; st > 0; st >>= 1) { if (t < st) red[t] += red[t + st]; __syncthreads(); }
    const float inv = rsqrtf(red[0] * (1.0f / 1024.0f) + 1e-5f);
    #pragma unroll
    for (int i = 0; i < 4; ++i) {
        const int d = t + 256 * i;
        out[row * 1024 + d] = (v[i] - mean) * inv * g[d] + b[d];
    }
}

// ================================================================ driver
extern "C" void kernel_launch(void* const* d_in, const int* in_sizes, int n_in,
                              void* d_out, int out_size, void* d_ws, size_t ws_size,
                              hipStream_t stream)
{
    (void)in_sizes; (void)n_in; (void)out_size; (void)ws_size;
    const int   Lc = 4;
    const long  D  = 1024, F = 4096, S = 1024, Bb = 2, Hh = 16;
    const long  MD = Bb * S;          // 2048 rows
    const long  SS = S * S;           // 1M

    const int*   ids = (const int*)  d_in[0];
    const float* emb = (const float*)d_in[1];
    const float* Wq  = (const float*)d_in[2];
    const float* bq  = (const float*)d_in[3];
    const float* Wk  = (const float*)d_in[4];
    const float* bk  = (const float*)d_in[5];
    const float* Wv  = (const float*)d_in[6];
    const float* bv  = (const float*)d_in[7];
    const float* Wo  = (const float*)d_in[8];
    const float* bo  = (const float*)d_in[9];
    const float* W1  = (const float*)d_in[10];
    const float* b1  = (const float*)d_in[11];
    const float* W2  = (const float*)d_in[12];
    const float* b2  = (const float*)d_in[13];
    const float* ln1g = (const float*)d_in[14];
    const float* ln1b = (const float*)d_in[15];
    const float* ln2g = (const float*)d_in[16];
    const float* ln2b = (const float*)d_in[17];

    float* out  = (float*)d_out;
    float* attn = out + MD * D;                  // [L,B,H,S,S] region
    float* ws   = (float*)d_ws;

    float* h   = ws;                   // 2M floats
    float* q   = ws + 2 * 1024 * 1024;
    float* k   = ws + 4 * 1024 * 1024;
    float* v   = ws + 6 * 1024 * 1024;
    float* ff1 = ws + 8 * 1024 * 1024; // 8M floats
    float* ctx = q;                    // reuse (q dead after scores)
    float* tmp = k;                    // reuse (k dead after scores)

    embed_pe<<<(int)MD, 256, 0, stream>>>(ids, emb, h);

    for (int l = 0; l < Lc; ++l) {
        const float* Wq_l = Wq + (size_t)l * D * D;
        const float* Wk_l = Wk + (size_t)l * D * D;
        const float* Wv_l = Wv + (size_t)l * D * D;
        const float* Wo_l = Wo + (size_t)l * D * D;
        const float* W1_l = W1 + (size_t)l * D * F;
        const float* W2_l = W2 + (size_t)l * F * D;
        float* attnL = attn + (size_t)l * Bb * Hh * SS;

        // Q, K, V projections: [2048,1024] @ [1024,1024]
        launch_gemm<false,false>(stream, h, Wq_l, bq + l * D, q, MD, D, D, D, D, D,
                                 1, 1, 0,0,0,0,0,0, 1.0f);
        launch_gemm<false,false>(stream, h, Wk_l, bk + l * D, k, MD, D, D, D, D, D,
                                 1, 1, 0,0,0,0,0,0, 1.0f);
        launch_gemm<false,false>(stream, h, Wv_l, bv + l * D, v, MD, D, D, D, D, D,
                                 1, 1, 0,0,0,0,0,0, 1.0f);

        // scores = Q_h @ K_h^T * 1/8  -> written straight into d_out attn slot
        launch_gemm<true,false>(stream, q, k, nullptr, attnL,
                                S, S, 64, D, D, S,
                                (int)(Bb * Hh), (int)Hh,
                                S * D, 64, S * D, 64, Hh * SS, SS, 0.125f);

        // softmax in place over each row of the attention matrices
        softmax_rows<<<(int)(Bb * Hh * S), 256, 0, stream>>>(attnL);

        // ctx = aw @ V_h   [1024,1024] @ [1024,64] per (b,h)
        launch_gemm<false,false>(stream, attnL, v, nullptr, ctx,
                                 S, 64, S, S, D, D,
                                 (int)(Bb * Hh), (int)Hh,
                                 Hh * SS, SS, S * D, 64, S * D, 64, 1.0f);

        // attn_out = ctx @ Wo + bo ; h = LN(h + attn_out)
        launch_gemm<false,false>(stream, ctx, Wo_l, bo + l * D, tmp, MD, D, D, D, D, D,
                                 1, 1, 0,0,0,0,0,0, 1.0f);
        add_layernorm<<<(int)MD, 256, 0, stream>>>(h, tmp, ln1g + l * D, ln1b + l * D, h);

        // FFN
        launch_gemm<false,true >(stream, h, W1_l, b1 + l * F, ff1, MD, F, D, D, F, F,
                                 1, 1, 0,0,0,0,0,0, 1.0f);
        launch_gemm<false,false>(stream, ff1, W2_l, b2 + l * D, tmp, MD, D, F, F, D, D,
                                 1, 1, 0,0,0,0,0,0, 1.0f);
        add_layernorm<<<(int)MD, 256, 0, stream>>>(h, tmp, ln2g + l * D, ln2b + l * D,
                                                   (l == Lc - 1) ? out : h);
    }
}
